// FlowEmbedding_18494129176627
// MI455X (gfx1250) — matrix operand order repaired and checked by reading, not compiled, not run on hardware
//
#include <hip/hip_runtime.h>
#include <hip/hip_bf16.h>

// ---------------------------------------------------------------------------
// FlowEmbedding on gfx1250 (MI455X).
//   - KNN (S=64 of 2048) per query via LDS iterative argmin (tie-break = lower
//     index, matching jax.lax.top_k on -dist).
//   - 3 pointwise-conv layers as bf16 WMMA GEMMs (v_wmma_f32_16x16x32_bf16),
//     fp32 accumulation. BN stats (biased var over B*N*S) accumulated from the
//     fp32 accumulators via LDS ds_add_f32 + one global atomic per channel per
//     workgroup. BN+ReLU folded into the next layer's A-staging as per-channel
//     scale/shift. Activations kept bf16 [P][128], updated in place.
// ---------------------------------------------------------------------------

#define BATCH 4
#define NPTS  2048
#define CHF   128
#define SNBR  64
#define PTOT  (BATCH*NPTS*SNBR)   // 524288 positions
#define K0RAW 259
#define K0PAD 288                 // 9 K-chunks of 32
#define APAD0 304                 // LDS row pad (ushort), 608B rows (16B aligned)
#define APAD1 136                 // LDS row pad (ushort), 272B rows (16B aligned)

typedef __bf16        bf16x16 __attribute__((ext_vector_type(16)));
typedef float         f32x8   __attribute__((ext_vector_type(8)));
typedef unsigned int  u32x8   __attribute__((ext_vector_type(8)));

__device__ __forceinline__ unsigned short f2bf(float f) {
  union { float f; unsigned u; } x; x.f = f;
  unsigned r = x.u + 0x7fffu + ((x.u >> 16) & 1u);   // round-to-nearest-even
  return (unsigned short)(r >> 16);
}
__device__ __forceinline__ float bf2f(unsigned short h) {
  union { unsigned u; float f; } x; x.u = ((unsigned)h) << 16;
  return x.f;
}
__device__ __forceinline__ f32x8 wmma_bf16(u32x8 a, u32x8 b, f32x8 c) {
  return __builtin_amdgcn_wmma_f32_16x16x32_bf16(
      false, __builtin_bit_cast(bf16x16, a),
      false, __builtin_bit_cast(bf16x16, b),
      (short)0, c, false, false);
}

// --------------------------- weight prep + stat zero ------------------------
__global__ __launch_bounds__(256) void prep_kernel(
    const float* __restrict__ W0, const float* __restrict__ W1,
    const float* __restrict__ W2,
    unsigned short* __restrict__ W0bf, unsigned short* __restrict__ W1bf,
    unsigned short* __restrict__ W2bf, float* __restrict__ stats) {
  const int stride = gridDim.x * blockDim.x;
  const int t0 = blockIdx.x * blockDim.x + threadIdx.x;
  for (int i = t0; i < CHF * K0PAD; i += stride) {
    int o = i / K0PAD, k = i - o * K0PAD;
    W0bf[i] = f2bf(k < K0RAW ? W0[o * K0RAW + k] : 0.0f);
  }
  for (int i = t0; i < CHF * CHF; i += stride) {
    W1bf[i] = f2bf(W1[i]);
    W2bf[i] = f2bf(W2[i]);
  }
  for (int i = t0; i < 6 * CHF; i += stride) stats[i] = 0.0f;
}

// --------------------------------- KNN -------------------------------------
__global__ __launch_bounds__(256) void knn_kernel(
    const float* __restrict__ xyz1, const float* __restrict__ xyz2,
    int* __restrict__ idxbuf) {
  __shared__ float sd[NPTS];
  __shared__ float rv[256];
  __shared__ int   ri[256];
  const int bn = blockIdx.x;          // b*NPTS + n1
  const int b  = bn >> 11;
  const int n1 = bn & (NPTS - 1);
  const int tid = threadIdx.x;

  const float qx = xyz1[(b * 3 + 0) * NPTS + n1];
  const float qy = xyz1[(b * 3 + 1) * NPTS + n1];
  const float qz = xyz1[(b * 3 + 2) * NPTS + n1];

  for (int j = tid; j < NPTS; j += 256) {
    float dx = xyz2[(b * 3 + 0) * NPTS + j] - qx;
    float dy = xyz2[(b * 3 + 1) * NPTS + j] - qy;
    float dz = xyz2[(b * 3 + 2) * NPTS + j] - qz;
    sd[j] = dx * dx + dy * dy + dz * dz;
  }
  __syncthreads();

  for (int it = 0; it < SNBR; ++it) {
    float bv = 3.4e38f;
    int   bi = NPTS;
    for (int j = tid; j < NPTS; j += 256) {
      float v = sd[j];
      if (v < bv || (v == bv && j < bi)) { bv = v; bi = j; }
    }
    rv[tid] = bv; ri[tid] = bi;
    __syncthreads();
    for (int off = 128; off > 0; off >>= 1) {
      if (tid < off) {
        float v2 = rv[tid + off]; int i2 = ri[tid + off];
        if (v2 < rv[tid] || (v2 == rv[tid] && i2 < ri[tid])) {
          rv[tid] = v2; ri[tid] = i2;
        }
      }
      __syncthreads();
    }
    if (tid == 0) {
      idxbuf[bn * SNBR + it] = ri[0];
      sd[ri[0]] = 3.4e38f;
    }
    __syncthreads();
  }
}

// ----------------------- layer 0: gather + concat + GEMM --------------------
__global__ __launch_bounds__(128) void gemm0_kernel(
    const float* __restrict__ xyz1, const float* __restrict__ xyz2,
    const float* __restrict__ feat1, const float* __restrict__ feat2,
    const unsigned short* __restrict__ W0bf, const float* __restrict__ bias,
    const int* __restrict__ idxbuf, unsigned short* __restrict__ y,
    float* __restrict__ gsum, float* __restrict__ gsumsq) {
  __shared__ unsigned short sA[4][16][APAD0];
  __shared__ float s_sum[CHF], s_sq[CHF];
  const int tid = threadIdx.x;
  if (tid < CHF) { s_sum[tid] = 0.0f; s_sq[tid] = 0.0f; }
  __syncthreads();

  const int wave = tid >> 5, lane = tid & 31;
  const int p0   = (blockIdx.x * 4 + wave) << 4;        // 16-row M tile
  const int b    = p0 / (NPTS * SNBR);
  const int rem  = p0 - b * (NPTS * SNBR);
  const int n1   = rem / SNBR;                          // shared by all 16 rows

  const float q0 = xyz1[(b * 3 + 0) * NPTS + n1];
  const float q1 = xyz1[(b * 3 + 1) * NPTS + n1];
  const float q2 = xyz1[(b * 3 + 2) * NPTS + n1];

  // Stage A tile [16][288] bf16 into LDS (gather + concat on the fly).
  for (int row = 0; row < 16; ++row) {
    const int idx = idxbuf[p0 + row];
#pragma unroll
    for (int j = 0; j < 9; ++j) {
      int c = lane + (j << 5);
      float v;
      if (c < 3) {
        float q = (c == 0) ? q0 : ((c == 1) ? q1 : q2);
        v = xyz2[(b * 3 + c) * NPTS + idx] - q;
      } else if (c < 131) {
        v = feat2[(b * CHF + (c - 3)) * NPTS + idx];
      } else if (c < K0RAW) {
        v = feat1[(b * CHF + (c - 131)) * NPTS + n1];
      } else {
        v = 0.0f;
      }
      sA[wave][row][c] = f2bf(v);
    }
  }

  f32x8 acc[8] = {};
  const unsigned int* rowp =
      (const unsigned int*)&sA[wave][lane & 15][0];
  const int hi = lane >> 4;                 // 0: lanes 0-15, 1: lanes 16-31

#pragma unroll
  for (int kk = 0; kk < 9; ++kk) {
    u32x8 af;
    {
      const uint4* rp4 = (const uint4*)(rowp + kk * 16 + (hi ? 4 : 0));
      uint4 a0 = rp4[0];        // K kb..kb+7
      uint4 a1 = rp4[2];        // K kb+16..kb+23
      af[0] = a0.x; af[1] = a0.y; af[2] = a0.z; af[3] = a0.w;
      af[4] = a1.x; af[5] = a1.y; af[6] = a1.z; af[7] = a1.w;
    }
#pragma unroll
    for (int nt = 0; nt < 8; ++nt) {
      const uint4* wp4 =
          (const uint4*)(W0bf + (nt * 16 + (lane & 15)) * K0PAD);
      uint4 b0v = wp4[kk * 4 + hi];
      uint4 b1v = wp4[kk * 4 + hi + 2];
      u32x8 bf;
      bf[0] = b0v.x; bf[1] = b0v.y; bf[2] = b0v.z; bf[3] = b0v.w;
      bf[4] = b1v.x; bf[5] = b1v.y; bf[6] = b1v.z; bf[7] = b1v.w;
      acc[nt] = wmma_bf16(af, bf, acc[nt]);
    }
  }

  // Epilogue: bias, BN-stat accumulation, bf16 store.
  const int n_lo = lane & 15;
#pragma unroll
  for (int nt = 0; nt < 8; ++nt) {
    const int ch = nt * 16 + n_lo;
    const float bb = bias[ch];
    float ps = 0.0f, pq = 0.0f;
#pragma unroll
    for (int r = 0; r < 8; ++r) {
      float v = acc[nt][r] + bb;
      ps += v; pq += v * v;
      const int m = r + (hi << 3);
      y[(p0 + m) * CHF + ch] = f2bf(v);
    }
    atomicAdd(&s_sum[ch], ps);
    atomicAdd(&s_sq[ch], pq);
  }
  __syncthreads();
  if (tid < CHF) {
    atomicAdd(&gsum[tid], s_sum[tid]);
    atomicAdd(&gsumsq[tid], s_sq[tid]);
  }
}

// ------------------- BN params (mean/var -> scale/shift) --------------------
__global__ __launch_bounds__(128) void bnparams_kernel(
    const float* __restrict__ g, const float* __restrict__ bt,
    const float* __restrict__ gsum, const float* __restrict__ gsumsq,
    float* __restrict__ scale, float* __restrict__ shift) {
  const int c = threadIdx.x;
  const float invP = 1.0f / (float)PTOT;
  float mean = gsum[c] * invP;
  float var  = fmaxf(gsumsq[c] * invP - mean * mean, 0.0f);
  float inv  = rsqrtf(var + 1e-3f);
  float sc = g[c] * inv;
  scale[c] = sc;
  shift[c] = bt[c] - mean * sc;
}

// ------------- layers 1/2: BN+ReLU(A) fused, in-place y, K=128 --------------
__global__ __launch_bounds__(128) void gemm12_kernel(
    const unsigned short* __restrict__ Wbf, const float* __restrict__ bias,
    const float* __restrict__ scale, const float* __restrict__ shift,
    unsigned short* __restrict__ y,
    float* __restrict__ gsum, float* __restrict__ gsumsq) {
  __shared__ unsigned short sA[4][16][APAD1];
  __shared__ float s_sum[CHF], s_sq[CHF];
  const int tid = threadIdx.x;
  if (tid < CHF) { s_sum[tid] = 0.0f; s_sq[tid] = 0.0f; }
  __syncthreads();

  const int wave = tid >> 5, lane = tid & 31;
  const int p0 = (blockIdx.x * 4 + wave) << 4;

  // Stage: A[row][c] = relu(bn(y_prev[row][c])) as bf16, full K before any
  // store -> safe in-place update of the same 16 rows afterwards.
  for (int row = 0; row < 16; ++row) {
    const unsigned int* yrow = (const unsigned int*)(y + (p0 + row) * CHF);
    unsigned int* drow = (unsigned int*)&sA[wave][row][0];
#pragma unroll
    for (int j = 0; j < 2; ++j) {
      int cu = lane + (j << 5);          // packs channels 2cu, 2cu+1
      unsigned int u = yrow[cu];
      int c0 = cu * 2;
      float f0 = fmaxf(fmaf(bf2f((unsigned short)(u & 0xffffu)),
                            scale[c0], shift[c0]), 0.0f);
      float f1 = fmaxf(fmaf(bf2f((unsigned short)(u >> 16)),
                            scale[c0 + 1], shift[c0 + 1]), 0.0f);
      drow[cu] = (unsigned)f2bf(f0) | ((unsigned)f2bf(f1) << 16);
    }
  }

  f32x8 acc[8] = {};
  const unsigned int* rowp = (const unsigned int*)&sA[wave][lane & 15][0];
  const int hi = lane >> 4;

#pragma unroll
  for (int kk = 0; kk < 4; ++kk) {
    u32x8 af;
    {
      const uint4* rp4 = (const uint4*)(rowp + kk * 16 + (hi ? 4 : 0));
      uint4 a0 = rp4[0];
      uint4 a1 = rp4[2];
      af[0] = a0.x; af[1] = a0.y; af[2] = a0.z; af[3] = a0.w;
      af[4] = a1.x; af[5] = a1.y; af[6] = a1.z; af[7] = a1.w;
    }
#pragma unroll
    for (int nt = 0; nt < 8; ++nt) {
      const uint4* wp4 = (const uint4*)(Wbf + (nt * 16 + (lane & 15)) * CHF);
      uint4 b0v = wp4[kk * 4 + hi];
      uint4 b1v = wp4[kk * 4 + hi + 2];
      u32x8 bf;
      bf[0] = b0v.x; bf[1] = b0v.y; bf[2] = b0v.z; bf[3] = b0v.w;
      bf[4] = b1v.x; bf[5] = b1v.y; bf[6] = b1v.z; bf[7] = b1v.w;
      acc[nt] = wmma_bf16(af, bf, acc[nt]);
    }
  }

  const int n_lo = lane & 15;
#pragma unroll
  for (int nt = 0; nt < 8; ++nt) {
    const int ch = nt * 16 + n_lo;
    const float bb = bias[ch];
    float ps = 0.0f, pq = 0.0f;
#pragma unroll
    for (int r = 0; r < 8; ++r) {
      float v = acc[nt][r] + bb;
      ps += v; pq += v * v;
      const int m = r + (hi << 3);
      y[(p0 + m) * CHF + ch] = f2bf(v);     // in-place overwrite
    }
    atomicAdd(&s_sum[ch], ps);
    atomicAdd(&s_sq[ch], pq);
  }
  __syncthreads();
  if (tid < CHF) {
    atomicAdd(&gsum[tid], s_sum[tid]);
    atomicAdd(&gsumsq[tid], s_sq[tid]);
  }
}

// ---------------- final BN+ReLU + max-pool over neighbors -------------------
__global__ __launch_bounds__(128) void finalize_kernel(
    const unsigned short* __restrict__ y, const float* __restrict__ scale,
    const float* __restrict__ shift, float* __restrict__ out) {
  const int bn = blockIdx.x;            // b*NPTS + n1
  const int c  = threadIdx.x;
  const float sc = scale[c], sh = shift[c];
  const unsigned short* base = y + (long)bn * SNBR * CHF + c;
  float m = -3.4e38f;
  for (int s = 0; s < SNBR; ++s) {
    float v = fmaxf(fmaf(bf2f(base[s * CHF]), sc, sh), 0.0f);
    m = fmaxf(m, v);
  }
  const int b = bn >> 11, n1 = bn & (NPTS - 1);
  out[(b * CHF + c) * NPTS + n1] = m;
}

// ----------------------------------------------------------------------------
extern "C" void kernel_launch(void* const* d_in, const int* in_sizes, int n_in,
                              void* d_out, int out_size, void* d_ws,
                              size_t ws_size, hipStream_t stream) {
  (void)in_sizes; (void)n_in; (void)out_size;
  const float* xyz1  = (const float*)d_in[0];
  const float* xyz2  = (const float*)d_in[1];
  const float* feat1 = (const float*)d_in[2];
  const float* feat2 = (const float*)d_in[3];
  const float* W0 = (const float*)d_in[4];
  const float* b0 = (const float*)d_in[5];
  const float* g0 = (const float*)d_in[6];
  const float* bt0 = (const float*)d_in[7];
  const float* W1 = (const float*)d_in[8];
  const float* b1 = (const float*)d_in[9];
  const float* g1 = (const float*)d_in[10];
  const float* bt1 = (const float*)d_in[11];
  const float* W2 = (const float*)d_in[12];
  const float* b2 = (const float*)d_in[13];
  const float* g2 = (const float*)d_in[14];
  const float* bt2 = (const float*)d_in[15];

  char* ws = (char*)d_ws;
  size_t off = 0;
  auto walloc = [&](size_t bytes) -> void* {
    void* p = ws + off;
    off = (off + bytes + 255) & ~(size_t)255;
    return p;
  };
  unsigned short* W0bf = (unsigned short*)walloc((size_t)CHF * K0PAD * 2);
  unsigned short* W1bf = (unsigned short*)walloc((size_t)CHF * CHF * 2);
  unsigned short* W2bf = (unsigned short*)walloc((size_t)CHF * CHF * 2);
  float* stats = (float*)walloc(6 * CHF * sizeof(float)); // sum/sq x 3 layers
  float* bnp   = (float*)walloc(6 * CHF * sizeof(float)); // scale/shift x 3
  int*   idxb  = (int*)walloc((size_t)BATCH * NPTS * SNBR * sizeof(int));
  unsigned short* y = (unsigned short*)walloc((size_t)PTOT * CHF * 2);
  if (off > ws_size) return;  // workspace too small; avoid OOB

  float* sum0 = stats + 0 * CHF; float* sq0 = stats + 1 * CHF;
  float* sum1 = stats + 2 * CHF; float* sq1 = stats + 3 * CHF;
  float* sum2 = stats + 4 * CHF; float* sq2 = stats + 5 * CHF;
  float* sc0 = bnp + 0 * CHF; float* sh0 = bnp + 1 * CHF;
  float* sc1 = bnp + 2 * CHF; float* sh1 = bnp + 3 * CHF;
  float* sc2 = bnp + 4 * CHF; float* sh2 = bnp + 5 * CHF;

  prep_kernel<<<64, 256, 0, stream>>>(W0, W1, W2, W0bf, W1bf, W2bf, stats);
  knn_kernel<<<BATCH * NPTS, 256, 0, stream>>>(xyz1, xyz2, idxb);
  gemm0_kernel<<<PTOT / 64, 128, 0, stream>>>(xyz1, xyz2, feat1, feat2, W0bf,
                                              b0, idxb, y, sum0, sq0);
  bnparams_kernel<<<1, 128, 0, stream>>>(g0, bt0, sum0, sq0, sc0, sh0);
  gemm12_kernel<<<PTOT / 64, 128, 0, stream>>>(W1bf, b1, sc0, sh0, y, sum1, sq1);
  bnparams_kernel<<<1, 128, 0, stream>>>(g1, bt1, sum1, sq1, sc1, sh1);
  gemm12_kernel<<<PTOT / 64, 128, 0, stream>>>(W2bf, b2, sc1, sh1, y, sum2, sq2);
  bnparams_kernel<<<1, 128, 0, stream>>>(g2, bt2, sum2, sq2, sc2, sh2);
  finalize_kernel<<<BATCH * NPTS, 128, 0, stream>>>(y, sc2, sh2, (float*)d_out);
}
